// Net_23587960389669
// MI455X (gfx1250) — matrix operand order repaired and checked by reading, not compiled, not run on hardware
//
#include <hip/hip_runtime.h>
#include <math.h>

// Swin shifted-window attention, fully fused, gfx1250 WMMA (wave32).
// B=16, H=W=112, C=256, NH=8, d=32, WS=7, SH=3, N=49. 4096 windows.

typedef __attribute__((ext_vector_type(16))) _Float16 v16h;
typedef __attribute__((ext_vector_type(4)))  _Float16 v4h;
typedef __attribute__((ext_vector_type(8)))  float    v8f;
typedef __attribute__((ext_vector_type(4)))  float    v4f;

#define CCH   256          // channels
#define NHEAD 8
#define DH    32
#define NTOK  49
#define HWDIM 112
#define LDH   264          // half leading dim for 64-row f16 tiles (bank-conflict pad)
#define LDP   66           // P (softmax probs) leading dim, halves (33 DW, coprime 64)
#define LDS_S 65           // S (scores) leading dim, floats (coprime with 64 banks)

__device__ __forceinline__ v8f wmma16(v16h a, v16h b, v8f c) {
  // D = A*B + C, f16 in / f32 acc  ->  v_wmma_f32_16x16x32_f16
  return __builtin_amdgcn_wmma_f32_16x16x32_f16(false, a, false, b, (short)0, c,
                                                false, false);
}

// A fragment (16x32 f16), source row-major MxK with leading dim ld.
// lane -> M = lane%16 ; lanes<16 hold K=0..15, lanes>=16 hold K=16..31
__device__ __forceinline__ v16h load_a_rm(const _Float16* base, int ld, int m0,
                                          int k0, int lane) {
  int m  = m0 + (lane & 15);
  int kg = (lane & 16);
  v16h a;
#pragma unroll
  for (int e = 0; e < 16; ++e) {
    int k = k0 + kg + e;
    a[e] = base[m * ld + k];
  }
  return a;
}

// B fragment (32x16 f16) from a matrix stored N x K row-major (i.e. W^T as B):
// lane -> N = lane%16 ; lanes<16 hold K=0..15, lanes>=16 hold K=16..31
__device__ __forceinline__ v16h load_b_nk(const _Float16* base, int ld, int k0,
                                          int n0, int lane) {
  int n  = n0 + (lane & 15);
  int kb = (lane & 16);
  v16h b;
#pragma unroll
  for (int e = 0; e < 16; ++e) {
    int k = k0 + kb + e;
    b[e] = base[n * ld + k];
  }
  return b;
}

// B fragment (32x16 f16) from a matrix stored K x N row-major.
__device__ __forceinline__ v16h load_b_kn(const _Float16* base, int ld, int k0,
                                          int n0, int lane) {
  int n  = n0 + (lane & 15);
  int kb = (lane & 16);
  v16h b;
#pragma unroll
  for (int e = 0; e < 16; ++e) {
    int k = k0 + kb + e;
    b[e] = base[k * ld + n];
  }
  return b;
}

// region id along one axis of the SHIFTED image (for the Swin attn mask)
__device__ __forceinline__ int region1d(int g) {
  // H-WS = 105, H-SH = 109
  return (g < 105) ? 0 : ((g < 109) ? 1 : 2);
}

__global__ void convert_weights_kernel(const float* __restrict__ qkvw,
                                       const float* __restrict__ projw,
                                       _Float16* __restrict__ wbuf) {
  int i = blockIdx.x * blockDim.x + threadIdx.x;
  if (i < 3 * CCH * CCH) wbuf[i] = (_Float16)qkvw[i];                  // 768x256
  if (i < CCH * CCH)     wbuf[3 * CCH * CCH + i] = (_Float16)projw[i]; // 256x256
}

__global__ __launch_bounds__(256)
void swin_window_attn_kernel(const float* __restrict__ input,
                             const _Float16* __restrict__ wbuf,
                             const float* __restrict__ qkv_bias,
                             const float* __restrict__ proj_bias,
                             const float* __restrict__ rpb,   // (NH,49,49)
                             float* __restrict__ out) {
  __shared__ _Float16 sXW[64 * LDH];   // window input (f16); reused as attn output O
  __shared__ _Float16 sQ[64 * LDH];    // Q (pre-scaled), token x (head*32+d)
  __shared__ _Float16 sK[64 * LDH];
  __shared__ _Float16 sV[64 * LDH];
  __shared__ float    sS[64 * LDS_S];  // attention scores (one head at a time)
  __shared__ _Float16 sP[64 * LDP];    // softmax probs f16

  const int tid  = threadIdx.x;
  const int wid  = tid >> 5;           // 0..7 (8 waves)
  const int lane = tid & 31;
  const int hi8  = (lane & 16) ? 8 : 0;

  const int blk = blockIdx.x;          // b*256 + wr*16 + wc
  const int b   = blk >> 8;
  const int wr  = (blk >> 4) & 15;
  const int wc  = blk & 15;

  const _Float16* qkvw  = wbuf;                   // 768 x 256, N x K
  const _Float16* projw = wbuf + 3 * CCH * CCH;   // 256 x 256, N x K

  // Warm L2/L0 with the shared f16 weights (global_prefetch_b8; cacheline grain)
  {
    const char* wb = (const char*)wbuf;
    const int wbytes = (3 * CCH * CCH + CCH * CCH) * 2;  // 512 KB
    for (int off = tid * 128; off < wbytes; off += 256 * 128)
      __builtin_prefetch(wb + off, 0, 3);
  }

  // ---------------- Phase 0: gather shifted window, f32 -> f16 into LDS -----
  // vectorized: one float4 (b128) load + one 64-bit LDS store per step
  for (int idx = tid; idx < 64 * (CCH / 4); idx += 256) {
    int row = idx >> 6;          // token (pad rows 49..63 -> 0)
    int ch4 = (idx & 63) << 2;   // channel group of 4
    v4f v = {0.0f, 0.0f, 0.0f, 0.0f};
    if (row < NTOK) {
      int gh = wr * 7 + row / 7;            // coords in shifted image
      int gw = wc * 7 + row % 7;
      int ih = gh + 3; if (ih >= HWDIM) ih -= HWDIM;   // roll(-3): read input[g+3]
      int iw = gw + 3; if (iw >= HWDIM) iw -= HWDIM;
      const float* src = input + (((size_t)b * HWDIM + ih) * HWDIM + iw) * CCH + ch4;
      v = *(const v4f*)src;
    }
    v4h h4;
    h4[0] = (_Float16)v[0]; h4[1] = (_Float16)v[1];
    h4[2] = (_Float16)v[2]; h4[3] = (_Float16)v[3];
    *(v4h*)&sXW[row * LDH + ch4] = h4;
  }
  __syncthreads();

  // ---------------- Phase 1: QKV GEMM  (64pad x 768) = XW(64x256) @ Wqkv^T ---
  const float qscale = 0.17677669529663687f;   // 32^-0.5
  for (int t = wid; t < 4 * 48; t += 8) {      // 4 M-tiles x 48 N-tiles
    int mt = t & 3, nt = t >> 2;
    v8f acc = {};
#pragma unroll
    for (int kb = 0; kb < 8; ++kb) {
      v16h a = load_a_rm(sXW, LDH, mt * 16, kb * 32, lane);
      v16h bfr = load_b_nk(qkvw, CCH, kb * 32, nt * 16, lane);
      acc = wmma16(a, bfr, acc);
    }
    int nglob = nt * 16 + (lane & 15);         // 0..767
    float bias = qkv_bias[nglob];
    int seg = nglob >> 8;                      // 0=Q 1=K 2=V
    int col = nglob & 255;
    _Float16* dst = (seg == 0) ? sQ : (seg == 1) ? sK : sV;
    float scl = (seg == 0) ? qscale : 1.0f;
#pragma unroll
    for (int r = 0; r < 8; ++r) {
      int row = mt * 16 + r + hi8;
      dst[row * LDH + col] = (_Float16)((acc[r] + bias) * scl);
    }
  }
  __syncthreads();

  // mask regions depend only on window position (nonzero only at wr/wc == 15)
  // ---------------- Phase 2: attention, one head at a time ------------------
  for (int h = 0; h < NHEAD; ++h) {
    // S = Qh (64x32) * Kh^T (32x64): 16 tiles, one K-step each
    for (int t = wid; t < 16; t += 8) {
      int mt = t & 3, nt = t >> 2;
      v8f acc = {};
      v16h a = load_a_rm(sQ, LDH, mt * 16, h * DH, lane);
      v16h bfr = load_b_nk(sK, LDH, h * DH, nt * 16, lane);  // K stored token x d
      acc = wmma16(a, bfr, acc);
      int j = nt * 16 + (lane & 15);
      int regj = 0, validj = (j < NTOK);
      if (validj) regj = region1d(wr * 7 + j / 7) * 3 + region1d(wc * 7 + j % 7);
#pragma unroll
      for (int r = 0; r < 8; ++r) {
        int i = mt * 16 + r + hi8;
        float v = -1.0e9f;
        if (i < NTOK && validj) {
          int regi = region1d(wr * 7 + i / 7) * 3 + region1d(wc * 7 + i % 7);
          float m = (regi != regj) ? -100.0f : 0.0f;
          v = acc[r] + rpb[((h * NTOK) + i) * NTOK + j] + m;
        }
        sS[i * LDS_S + j] = v;
      }
    }
    __syncthreads();

    // row softmax (64 rows, one thread per row; LDS_S=65 -> conflict-free)
    if (tid < 64) {
      float mx = -3.0e38f;
#pragma unroll 4
      for (int j = 0; j < 64; ++j) mx = fmaxf(mx, sS[tid * LDS_S + j]);
      float sum = 0.0f;
#pragma unroll 4
      for (int j = 0; j < 64; ++j) {
        float e = __expf(sS[tid * LDS_S + j] - mx);
        sS[tid * LDS_S + j] = e;
        sum += e;
      }
      float inv = 1.0f / sum;
#pragma unroll 4
      for (int j = 0; j < 64; ++j)
        sP[tid * LDP + j] = (_Float16)(sS[tid * LDS_S + j] * inv);
    }
    __syncthreads();

    // O = P (64x64) * Vh (64x32): 8 tiles, two K-steps; write into sXW
    for (int t = wid; t < 8; t += 8) {
      int mt = t & 3, nt = t >> 2;                 // nt in 0..1
      v8f acc = {};
#pragma unroll
      for (int kb = 0; kb < 2; ++kb) {
        v16h a = load_a_rm(sP, LDP, mt * 16, kb * 32, lane);
        v16h bfr = load_b_kn(sV, LDH, kb * 32, h * DH + nt * 16, lane);
        acc = wmma16(a, bfr, acc);
      }
      int col = h * DH + nt * 16 + (lane & 15);
#pragma unroll
      for (int r = 0; r < 8; ++r) {
        int row = mt * 16 + r + hi8;
        sXW[row * LDH + col] = (_Float16)acc[r];
      }
    }
    __syncthreads();   // protects sS/sP reuse next head
  }

  // ---------------- Phase 3: proj GEMM + un-partition + reverse shift -------
  for (int t = wid; t < 4 * 16; t += 8) {
    int mt = t & 3, nt = t >> 2;
    v8f acc = {};
#pragma unroll
    for (int kb = 0; kb < 8; ++kb) {
      v16h a = load_a_rm(sXW, LDH, mt * 16, kb * 32, lane);
      v16h bfr = load_b_nk(projw, CCH, kb * 32, nt * 16, lane);
      acc = wmma16(a, bfr, acc);
    }
    int ch = nt * 16 + (lane & 15);
    float pb = proj_bias[ch];
#pragma unroll
    for (int r = 0; r < 8; ++r) {
      int i = mt * 16 + r + hi8;
      if (i < NTOK) {
        int gh = wr * 7 + i / 7;
        int gw = wc * 7 + i % 7;
        int oh = gh + 3; if (oh >= HWDIM) oh -= HWDIM;  // roll(+3): write out[g+3]
        int ow = gw + 3; if (ow >= HWDIM) ow -= HWDIM;
        out[(((size_t)b * HWDIM + oh) * HWDIM + ow) * CCH + ch] = acc[r] + pb;
      }
    }
  }
}

extern "C" void kernel_launch(void* const* d_in, const int* in_sizes, int n_in,
                              void* d_out, int out_size, void* d_ws, size_t ws_size,
                              hipStream_t stream) {
  const float* input     = (const float*)d_in[0];
  const float* qkv_w     = (const float*)d_in[1];
  const float* qkv_b     = (const float*)d_in[2];
  const float* proj_w    = (const float*)d_in[3];
  const float* proj_b    = (const float*)d_in[4];
  const float* rpb       = (const float*)d_in[5];
  float* out             = (float*)d_out;
  _Float16* wbuf         = (_Float16*)d_ws;   // 768*256 + 256*256 halves = 512 KB

  int nconv = 3 * CCH * CCH;
  convert_weights_kernel<<<(nconv + 255) / 256, 256, 0, stream>>>(qkv_w, proj_w, wbuf);

  dim3 grid(16 * 16 * 16);   // B * (16x16 windows) = 4096
  dim3 block(256);           // 8 waves (wave32)
  swin_window_attn_kernel<<<grid, block, 0, stream>>>(input, wbuf, qkv_b, proj_b,
                                                      rpb, out);
}
